// MultiheadAttention_74380243632417
// MI455X (gfx1250) — matrix operand order repaired and checked by reading, not compiled, not run on hardware
//
#include <hip/hip_runtime.h>

// MHA: B=2, S=4096, D=512, H=8, HD=64  (fp32 in/out; bf16 WMMA internally)
#define Bb 2
#define Ss 4096
#define Dd 512
#define Hh 8
#define HDim 64

typedef __attribute__((ext_vector_type(16))) __bf16 v16bf;
typedef __attribute__((ext_vector_type(8)))  __bf16 v8bf;
typedef __attribute__((ext_vector_type(8)))  float  v8f;
typedef __attribute__((ext_vector_type(4)))  int    v4i;

static __device__ inline v16bf make16(v8bf lo, v8bf hi) {
  v16bf r;
#pragma unroll
  for (int i = 0; i < 8; ++i) { r[i] = lo[i]; r[8 + i] = hi[i]; }
  return r;
}

// A-fragment (16x32 bf16, ISA 7.12.2): lane L holds row M=L&15.
// lanes 0-15: K = k0+{0..7, 16..23}; lanes 16-31: K = k0+{8..15, 24..31}.
static __device__ inline v16bf load_a_frag(const __bf16* base, int ld, int k0) {
  int lane = threadIdx.x & 31;
  int row  = lane & 15;
  int kk   = k0 + ((lane & 16) ? 8 : 0);
  const __bf16* p = base + (size_t)row * ld + kk;
  v8bf lo = *(const v8bf*)(p);
  v8bf hi = *(const v8bf*)(p + 16);
  return make16(lo, hi);
}

// B-fragment (32x16 bf16): lane L holds column N=L&15.
// lanes 0-15: K = k0+{0..7, 16..23}; lanes 16-31: K = k0+{8..15, 24..31}.
// 'bt' is B stored transposed (row = N, contiguous K), leading dim ld.
static __device__ inline v16bf load_b_frag(const __bf16* bt, int ld, int k0) {
  int lane = threadIdx.x & 31;
  int col  = lane & 15;
  int kk   = k0 + ((lane & 16) ? 8 : 0);
  const __bf16* p = bt + (size_t)col * ld + kk;
  v8bf lo = *(const v8bf*)(p);
  v8bf hi = *(const v8bf*)(p + 16);
  return make16(lo, hi);
}

static __device__ inline v8f wmma_bf16(v16bf a, v16bf b, v8f c) {
  return __builtin_amdgcn_wmma_f32_16x16x32_bf16(false, a, false, b, (short)0, c,
                                                 false, false);
}

// 16-byte global -> LDS copy; prefer the gfx1250 async-to-LDS path (ASYNCcnt).
static __device__ inline void async_copy16(const __bf16* g, __bf16* l) {
#if __has_builtin(__builtin_amdgcn_global_load_async_to_lds_b128)
  typedef __attribute__((address_space(1))) v4i* gp_t;  // pointer to AS1 int4
  typedef __attribute__((address_space(3))) v4i* lp_t;  // pointer to AS3 int4
  __builtin_amdgcn_global_load_async_to_lds_b128((gp_t)g, (lp_t)l, 0, 0);
#else
  *(v8bf*)l = *(const v8bf*)g;
#endif
}

static __device__ inline void wait_async_done() {
#if __has_builtin(__builtin_amdgcn_s_wait_asynccnt)
  __builtin_amdgcn_s_wait_asynccnt(0);
#else
  asm volatile("s_wait_asynccnt 0x0" ::: "memory");
#endif
}

// ---------------- conversion kernels ----------------
__global__ void cvt_f32_to_bf16(const float* __restrict__ src,
                                __bf16* __restrict__ dst, int n) {
  int i = blockIdx.x * 256 + threadIdx.x;
  if (i < n) dst[i] = (__bf16)src[i];
}

// wt[n*D + k] = w[k*D + n]  (store transposed so B-frags load contiguously)
__global__ void cvt_transpose_w(const float* __restrict__ w,
                                __bf16* __restrict__ wt) {
  int i = blockIdx.x * 256 + threadIdx.x;  // over D*D
  int k = i >> 9;          // / 512
  int n = i & (Dd - 1);    // % 512
  wt[(size_t)n * Dd + k] = (__bf16)w[i];
}

// ---------------- fused QKV projection ----------------
// z=0: Q (scaled by log2e/sqrt(HD)) -> [B,H,S,HD]; z=1: K -> [B,H,S,HD];
// z=2: V -> transposed [B,H,HD,S].
__global__ void __launch_bounds__(256)
qkv_gemm(const __bf16* __restrict__ Xb,
         const __bf16* __restrict__ Wqt, const __bf16* __restrict__ Wkt,
         const __bf16* __restrict__ Wvt,
         const float* __restrict__ bq, const float* __restrict__ bk,
         const float* __restrict__ bv,
         __bf16* __restrict__ Qb, __bf16* __restrict__ Kb,
         __bf16* __restrict__ Vtb) {
  int wave = threadIdx.x >> 5;
  int lane = threadIdx.x & 31;
  int tile = blockIdx.x * 8 + wave;   // 0..16383
  int mt = tile >> 5;                 // M tile (512)
  int nt = tile & 31;                 // N tile (32)
  int z  = blockIdx.y;

  const __bf16* Wt   = (z == 0) ? Wqt : (z == 1) ? Wkt : Wvt;
  const float*  bias = (z == 0) ? bq  : (z == 1) ? bk  : bv;

  const __bf16* Abase = Xb + (size_t)mt * 16 * Dd;
  const __bf16* Bbase = Wt + (size_t)nt * 16 * Dd;

  v8f acc = {};
#pragma unroll 4
  for (int k0 = 0; k0 < Dd; k0 += 32) {
    v16bf a = load_a_frag(Abase, Dd, k0);
    v16bf b = load_b_frag(Bbase, Dd, k0);
    acc = wmma_bf16(a, b, acc);
  }

  int col   = lane & 15;
  int half8 = (lane & 16) ? 8 : 0;
  int n     = nt * 16 + col;
  float bia = bias[n];
  int h  = n >> 6;
  int hd = n & (HDim - 1);
  const float qscale = 0.125f * 1.44269504f;  // 1/sqrt(64) * log2(e)

#pragma unroll
  for (int i = 0; i < 8; ++i) {
    int gs = mt * 16 + half8 + i;       // row in [0, B*S)
    int bI = gs >> 12;
    int s  = gs & (Ss - 1);
    float val = acc[i] + bia;
    if (z == 0) {
      Qb[(((size_t)(bI * Hh + h)) * Ss + s) * HDim + hd] = (__bf16)(val * qscale);
    } else if (z == 1) {
      Kb[(((size_t)(bI * Hh + h)) * Ss + s) * HDim + hd] = (__bf16)val;
    } else {
      Vtb[(((size_t)(bI * Hh + h)) * HDim + hd) * Ss + s] = (__bf16)val;
    }
  }
}

// ---------------- flash attention ----------------
// Block = 8 waves = 128 query rows of one (b,h). Keys in chunks of 64:
// K/V tiles staged in LDS once per block via async-to-LDS, shared by all waves.
__global__ void __launch_bounds__(256)
flash_attn(const __bf16* __restrict__ Qb, const __bf16* __restrict__ Kb,
           const __bf16* __restrict__ Vtb, __bf16* __restrict__ Attn) {
  __shared__ __attribute__((aligned(16))) __bf16 ldsK[64 * HDim];     // [key][hd]
  __shared__ __attribute__((aligned(16))) __bf16 ldsV[HDim * 64];     // [hd][key]
  __shared__ __attribute__((aligned(16))) __bf16 ldsP[8 * 16 * 64];   // per-wave P
  int wave = threadIdx.x >> 5;
  int lane = threadIdx.x & 31;
  int tid  = threadIdx.x;
  int bh = blockIdx.x >> 5;           // 16 (b,h) pairs
  int mb = blockIdx.x & 31;           // 32 query blocks of 128 rows
  int m0 = mb * 128 + wave * 16;

  const __bf16* Qh = Qb  + (size_t)bh * Ss * HDim;
  const __bf16* Kh = Kb  + (size_t)bh * Ss * HDim;
  const __bf16* Vh = Vtb + (size_t)bh * HDim * Ss;

  v16bf aQ0 = load_a_frag(Qh + (size_t)m0 * HDim, HDim, 0);
  v16bf aQ1 = load_a_frag(Qh + (size_t)m0 * HDim, HDim, 32);

  v8f o0 = {}, o1 = {}, o2 = {}, o3 = {};
  float mrow[8], lrow[8];
#pragma unroll
  for (int i = 0; i < 8; ++i) { mrow[i] = -1e30f; lrow[i] = 0.f; }

  __bf16* pw = ldsP + wave * (16 * 64);
  int col   = lane & 15;
  int half8 = (lane & 16) ? 8 : 0;
  int vr = tid >> 3, vs = (tid & 7) * 8;   // V-tile copy coords

  for (int kc = 0; kc < Ss; kc += 64) {
    // stage K chunk (contiguous 8KB) and V chunk (64 rows x 128B)
    async_copy16(Kh + (size_t)kc * HDim + tid * 8,        ldsK + tid * 8);
    async_copy16(Kh + (size_t)kc * HDim + 2048 + tid * 8, ldsK + 2048 + tid * 8);
    async_copy16(Vh + (size_t)vr * Ss + kc + vs,          ldsV + vr * 64 + vs);
    async_copy16(Vh + (size_t)(vr + 32) * Ss + kc + vs,   ldsV + (vr + 32) * 64 + vs);
    wait_async_done();
    __syncthreads();

    // scores: Q(16x64) @ K^T(64x64); scale folded into Q (log2 domain)
    v8f s0 = {}, s1 = {}, s2 = {}, s3 = {};
    s0 = wmma_bf16(aQ0, load_b_frag(ldsK +  0 * HDim, HDim, 0),  s0);
    s0 = wmma_bf16(aQ1, load_b_frag(ldsK +  0 * HDim, HDim, 32), s0);
    s1 = wmma_bf16(aQ0, load_b_frag(ldsK + 16 * HDim, HDim, 0),  s1);
    s1 = wmma_bf16(aQ1, load_b_frag(ldsK + 16 * HDim, HDim, 32), s1);
    s2 = wmma_bf16(aQ0, load_b_frag(ldsK + 32 * HDim, HDim, 0),  s2);
    s2 = wmma_bf16(aQ1, load_b_frag(ldsK + 32 * HDim, HDim, 32), s2);
    s3 = wmma_bf16(aQ0, load_b_frag(ldsK + 48 * HDim, HDim, 0),  s3);
    s3 = wmma_bf16(aQ1, load_b_frag(ldsK + 48 * HDim, HDim, 32), s3);

    // online softmax (base-2); C layout: VGPR i = row half8+i, cols over 16 lanes
#pragma unroll
    for (int i = 0; i < 8; ++i) {
      float mx = fmaxf(fmaxf(s0[i], s1[i]), fmaxf(s2[i], s3[i]));
#pragma unroll
      for (int msk = 1; msk < 16; msk <<= 1)
        mx = fmaxf(mx, __shfl_xor(mx, msk, 32));
      float mnew = fmaxf(mrow[i], mx);
      float corr = exp2f(mrow[i] - mnew);
      float p0 = exp2f(s0[i] - mnew);
      float p1 = exp2f(s1[i] - mnew);
      float p2 = exp2f(s2[i] - mnew);
      float p3 = exp2f(s3[i] - mnew);
      float rs = (p0 + p1) + (p2 + p3);
#pragma unroll
      for (int msk = 1; msk < 16; msk <<= 1)
        rs += __shfl_xor(rs, msk, 32);
      lrow[i] = lrow[i] * corr + rs;
      mrow[i] = mnew;
      o0[i] *= corr; o1[i] *= corr; o2[i] *= corr; o3[i] *= corr;
      int r = half8 + i;
      pw[r * 64 +  0 + col] = (__bf16)p0;
      pw[r * 64 + 16 + col] = (__bf16)p1;
      pw[r * 64 + 32 + col] = (__bf16)p2;
      pw[r * 64 + 48 + col] = (__bf16)p3;
    }
    asm volatile("s_wait_dscnt 0" ::: "memory");  // wave-local P RAW

    v16bf aP0 = load_a_frag(pw, 64, 0);   // P(16x64) in A layout
    v16bf aP1 = load_a_frag(pw, 64, 32);
    o0 = wmma_bf16(aP0, load_b_frag(ldsV +  0 * 64, 64, 0),  o0);
    o0 = wmma_bf16(aP1, load_b_frag(ldsV +  0 * 64, 64, 32), o0);
    o1 = wmma_bf16(aP0, load_b_frag(ldsV + 16 * 64, 64, 0),  o1);
    o1 = wmma_bf16(aP1, load_b_frag(ldsV + 16 * 64, 64, 32), o1);
    o2 = wmma_bf16(aP0, load_b_frag(ldsV + 32 * 64, 64, 0),  o2);
    o2 = wmma_bf16(aP1, load_b_frag(ldsV + 32 * 64, 64, 32), o2);
    o3 = wmma_bf16(aP0, load_b_frag(ldsV + 48 * 64, 64, 0),  o3);
    o3 = wmma_bf16(aP1, load_b_frag(ldsV + 48 * 64, 64, 32), o3);
    __syncthreads();   // all waves done with ldsK/ldsV before next chunk
  }

  // epilogue: O/l, merge heads -> Attn[B*S, D]
  int bI = bh >> 3, h = bh & 7;
#pragma unroll
  for (int i = 0; i < 8; ++i) {
    int s = m0 + half8 + i;
    float inv = 1.0f / lrow[i];
    size_t rowbase = ((size_t)(bI * Ss + s)) * Dd + h * HDim;
    Attn[rowbase +  0 + col] = (__bf16)(o0[i] * inv);
    Attn[rowbase + 16 + col] = (__bf16)(o1[i] * inv);
    Attn[rowbase + 32 + col] = (__bf16)(o2[i] * inv);
    Attn[rowbase + 48 + col] = (__bf16)(o3[i] * inv);
  }
}

// ---------------- output projection ----------------
__global__ void __launch_bounds__(256)
out_gemm(const __bf16* __restrict__ Ab, const __bf16* __restrict__ Wot,
         const float* __restrict__ bo, float* __restrict__ out) {
  int wave = threadIdx.x >> 5;
  int lane = threadIdx.x & 31;
  int tile = blockIdx.x * 8 + wave;
  int mt = tile >> 5;
  int nt = tile & 31;

  const __bf16* Abase = Ab  + (size_t)mt * 16 * Dd;
  const __bf16* Bbase = Wot + (size_t)nt * 16 * Dd;

  v8f acc = {};
#pragma unroll 4
  for (int k0 = 0; k0 < Dd; k0 += 32) {
    v16bf a = load_a_frag(Abase, Dd, k0);
    v16bf b = load_b_frag(Bbase, Dd, k0);
    acc = wmma_bf16(a, b, acc);
  }

  int col   = lane & 15;
  int half8 = (lane & 16) ? 8 : 0;
  int n     = nt * 16 + col;
  float bia = bo[n];
#pragma unroll
  for (int i = 0; i < 8; ++i) {
    int gs = mt * 16 + half8 + i;
    out[(size_t)gs * Dd + n] = acc[i] + bia;
  }
}

// ---------------- host side ----------------
extern "C" void kernel_launch(void* const* d_in, const int* in_sizes, int n_in,
                              void* d_out, int out_size, void* d_ws, size_t ws_size,
                              hipStream_t stream) {
  const float* x  = (const float*)d_in[0];
  const float* Wq = (const float*)d_in[1];
  const float* bq = (const float*)d_in[2];
  const float* Wk = (const float*)d_in[3];
  const float* bk = (const float*)d_in[4];
  const float* Wv = (const float*)d_in[5];
  const float* bv = (const float*)d_in[6];
  const float* Wo = (const float*)d_in[7];
  const float* bo = (const float*)d_in[8];
  float* out = (float*)d_out;

  const size_t nX = (size_t)Bb * Ss * Dd;   // 4,194,304 elems
  const size_t nW = (size_t)Dd * Dd;        //   262,144 elems

  __bf16* Xb   = (__bf16*)d_ws;
  __bf16* Wqt  = Xb  + nX;
  __bf16* Wkt  = Wqt + nW;
  __bf16* Wvt  = Wkt + nW;
  __bf16* Wot  = Wvt + nW;
  __bf16* Qb   = Wot + nW;
  __bf16* Kb   = Qb  + nX;
  __bf16* Vtb  = Kb  + nX;
  __bf16* Attn = Vtb + nX;   // total ~42 MB of d_ws

  cvt_f32_to_bf16<<<(int)(nX / 256), 256, 0, stream>>>(x, Xb, (int)nX);
  cvt_transpose_w<<<(int)(nW / 256), 256, 0, stream>>>(Wq, Wqt);
  cvt_transpose_w<<<(int)(nW / 256), 256, 0, stream>>>(Wk, Wkt);
  cvt_transpose_w<<<(int)(nW / 256), 256, 0, stream>>>(Wv, Wvt);
  cvt_transpose_w<<<(int)(nW / 256), 256, 0, stream>>>(Wo, Wot);

  qkv_gemm<<<dim3(2048, 3), 256, 0, stream>>>(Xb, Wqt, Wkt, Wvt, bq, bk, bv,
                                              Qb, Kb, Vtb);

  // B*H * (S/128) = 512 blocks, 8 waves each
  flash_attn<<<512, 256, 0, stream>>>(Qb, Kb, Vtb, Attn);

  out_gemm<<<2048, 256, 0, stream>>>(Attn, Wot, bo, out);

  (void)in_sizes; (void)n_in; (void)out_size; (void)ws_size;
}
